// AttentionCircuit_57423712747574
// MI455X (gfx1250) — matrix operand order repaired and checked by reading, not compiled, not run on hardware
//
#include <hip/hip_runtime.h>
#include <hip/hip_bf16.h>

// ---------------- problem constants ----------------
#define B_   2
#define S_   1024
#define D_   1024
#define H_   16
#define DH_  64
#define R_   128
#define M_   8
#define NN_  256
#define BS_  (B_ * S_)   // 2048

// ---------------- WMMA types ----------------
typedef __attribute__((ext_vector_type(16))) __bf16 v16bf;
typedef __attribute__((ext_vector_type(8)))  float  v8f;
typedef __attribute__((ext_vector_type(4)))  float  v4f;
typedef __attribute__((ext_vector_type(4)))  unsigned int v4u;

union FragBF { v16bf v; __bf16 h[16]; v4u q[2]; };
union FragF  { v8f  v; float  f[8]; };

__device__ __forceinline__ v8f wmma_bf16(v16bf a, v16bf b, v8f c) {
  return __builtin_amdgcn_wmma_f32_16x16x32_bf16(
      /*neg_a=*/false, a, /*neg_b=*/false, b,
      /*c_mod=*/(short)0, c, /*reuse_a=*/false, /*reuse_b=*/false);
}

// A-fragment (16x32 bf16, row-major A, ld in elements). CDNA5 layout:
// lane&15 = row; lanes 0-15 hold K {0..7,16..23}+k0; lanes 16-31 hold {8..15,24..31}+k0.
__device__ __forceinline__ FragBF load_a_frag(const __bf16* base, int ld, int row0, int k0) {
  int lane = threadIdx.x & 31;
  const __bf16* p = base + (size_t)(row0 + (lane & 15)) * ld + k0 + ((lane >> 4) << 3);
  FragBF f;
  f.q[0] = *(const v4u*)p;        // K = kh+0..7
  f.q[1] = *(const v4u*)(p + 16); // K = kh+16..23
  return f;
}

// B-fragment (32x16 bf16) from a transposed matrix BT[N][K] (row-major, ld in elements).
// lane&15 = column N; lanes 0-15 hold K k0+{0..7,16..23}, lanes 16-31 hold K k0+{8..15,24..31}.
__device__ __forceinline__ FragBF load_b_fragT(const __bf16* bt, int ld, int n0, int k0) {
  int lane = threadIdx.x & 31;
  const __bf16* p = bt + (size_t)(n0 + (lane & 15)) * ld + k0 + ((lane >> 4) << 3);
  FragBF f;
  f.q[0] = *(const v4u*)p;
  f.q[1] = *(const v4u*)(p + 16);
  return f;
}

// ---------------- conversion kernels ----------------
__global__ void k_f2bf(const float* __restrict__ src, __bf16* __restrict__ dst, int n) {
  int i = blockIdx.x * 256 + threadIdx.x;
  if (i < n) dst[i] = (__bf16)src[i];
}

// src[b][r][c] (f32) -> dst[b][c][r] (bf16)
__global__ void k_f2bf_t(const float* __restrict__ src, __bf16* __restrict__ dst,
                         int batch, int rows, int cols) {
  int i = blockIdx.x * 256 + threadIdx.x;
  int total = batch * rows * cols;
  if (i >= total) return;
  int b = i / (rows * cols);
  int rem = i - b * (rows * cols);
  int r = rem / cols, c = rem - r * cols;
  dst[(size_t)b * rows * cols + (size_t)c * rows + r] = (__bf16)src[i];
}

// V [BS_,D_] (heads in columns) -> Vt [(b*H+h)*DH + d][S]
__global__ void k_vtrans(const __bf16* __restrict__ Vb, __bf16* __restrict__ Vt) {
  int i = blockIdx.x * 256 + threadIdx.x;  // over B*H*DH*S
  int t = i & (S_ - 1);
  int rest = i >> 10;
  int d = rest & (DH_ - 1);
  int bh = rest >> 6;
  int b = bh >> 4, h = bh & 15;
  Vt[i] = Vb[(size_t)(b * S_ + t) * D_ + h * DH_ + d];
}

// --------- generic bf16 WMMA GEMM: C[M][N] = A[M][K] * BT[N][K]^T -----------
// one wave = 16x64 strip (1 A-frag, 4 B-frags, 4 WMMAs per k-step), pipelined.
__global__ void k_gemm(const __bf16* __restrict__ A, const __bf16* __restrict__ BT,
                       float* __restrict__ C, int Nc, int Kd,
                       long long sBT, long long sC) {
  int z = blockIdx.z;
  BT += (size_t)z * sBT;
  C  += (size_t)z * sC;
  int wid = threadIdx.x >> 5, lane = threadIdx.x & 31;
  int mt = blockIdx.y * 4 + wid;
  int nt0 = blockIdx.x * 4;
  v8f acc[4];
  v8f vz = {};
#pragma unroll
  for (int j = 0; j < 4; ++j) acc[j] = vz;

  FragBF a_cur = load_a_frag(A, Kd, mt * 16, 0);
  FragBF b_cur[4];
#pragma unroll
  for (int j = 0; j < 4; ++j) b_cur[j] = load_b_fragT(BT, Kd, (nt0 + j) * 16, 0);

  for (int k = 32; k < Kd; k += 32) {
    // issue next iteration's loads before consuming current fragments
    FragBF a_nxt = load_a_frag(A, Kd, mt * 16, k);
    FragBF b_nxt[4];
#pragma unroll
    for (int j = 0; j < 4; ++j) b_nxt[j] = load_b_fragT(BT, Kd, (nt0 + j) * 16, k);
    if (k + 32 < Kd)
      __builtin_prefetch(A + (size_t)(mt * 16 + (lane & 15)) * Kd + k + 32, 0, 3);
#pragma unroll
    for (int j = 0; j < 4; ++j) acc[j] = wmma_bf16(a_cur.v, b_cur[j].v, acc[j]);
    a_cur = a_nxt;
#pragma unroll
    for (int j = 0; j < 4; ++j) b_cur[j] = b_nxt[j];
  }
#pragma unroll
  for (int j = 0; j < 4; ++j) acc[j] = wmma_bf16(a_cur.v, b_cur[j].v, acc[j]);

  int n = lane & 15, mh = (lane >> 4) * 8;
#pragma unroll
  for (int j = 0; j < 4; ++j) {
    FragF cf; cf.v = acc[j];
#pragma unroll
    for (int v = 0; v < 8; ++v)
      C[(size_t)(mt * 16 + v + mh) * Nc + (nt0 + j) * 16 + n] = cf.f[v];
  }
}

// ---------------- gate softmax + group(32) + renormalize ----------------
__global__ void k_gates(const float* __restrict__ logits, float* __restrict__ gates) {
  int wid = threadIdx.x >> 5, lane = threadIdx.x & 31;
  int row = blockIdx.x * 4 + wid;
  const float* lp = logits + (size_t)row * NN_ + lane * 8;
  float x[8];
  float mx = -3.0e38f;
#pragma unroll
  for (int i = 0; i < 8; ++i) { x[i] = lp[i]; mx = fmaxf(mx, x[i]); }
  mx = fmaxf(mx, __shfl_xor(mx, 1, 32));
  mx = fmaxf(mx, __shfl_xor(mx, 2, 32));
  mx = fmaxf(mx, __shfl_xor(mx, 4, 32));
  mx = fmaxf(mx, __shfl_xor(mx, 8, 32));
  mx = fmaxf(mx, __shfl_xor(mx, 16, 32));
  float s = 0.f;
#pragma unroll
  for (int i = 0; i < 8; ++i) { x[i] = __expf(x[i] - mx); s += x[i]; }
  float grp = s + __shfl_xor(s, 1, 32);
  grp += __shfl_xor(grp, 2, 32);
  float tot = grp;
  tot += __shfl_xor(tot, 4, 32);
  tot += __shfl_xor(tot, 8, 32);
  tot += __shfl_xor(tot, 16, 32);
  if ((lane & 3) == 0)
    gates[(size_t)row * M_ + (lane >> 2)] = grp / (tot + 1e-8f * tot);
}

// ---------------- h = sum_m g[:,m] * allh[m] ----------------
__global__ void k_h(const float* __restrict__ allh, const float* __restrict__ g,
                    __bf16* __restrict__ hout) {
  int i = blockIdx.x * 256 + threadIdx.x;  // < BS_*R_
  int row = i >> 7, rr = i & 127;
  float s = 0.f;
#pragma unroll
  for (int m = 0; m < M_; ++m)
    s += g[(size_t)row * M_ + m] * allh[((size_t)m * BS_ + row) * R_ + rr];
  hout[i] = (__bf16)s;
}

// -------- restore: out = sum_m diag(g_m) * (h @ rT[m]^T), gate folded post-WMMA --------
__global__ void k_restore(const __bf16* __restrict__ hmat, const float* __restrict__ g,
                          const __bf16* __restrict__ rT, __bf16* __restrict__ out) {
  int wid = threadIdx.x >> 5, lane = threadIdx.x & 31;
  int mt = blockIdx.y * 4 + wid;   // 0..127
  int nt = blockIdx.x;             // 0..63
  int n = lane & 15, mh = (lane >> 4) * 8;
  FragBF afr[4];
#pragma unroll
  for (int kk = 0; kk < 4; ++kk) afr[kk] = load_a_frag(hmat, R_, mt * 16, kk * 32);
  // gates for this lane's 8 output rows (C-layout rows v+mh)
  float gv[8][8];
#pragma unroll
  for (int v = 0; v < 8; ++v) {
    const float* gp = g + (size_t)(mt * 16 + v + mh) * M_;
    v4f g0 = *(const v4f*)gp;
    v4f g1 = *(const v4f*)(gp + 4);
#pragma unroll
    for (int m = 0; m < 4; ++m) { gv[v][m] = g0[m]; gv[v][4 + m] = g1[m]; }
  }
  FragF accT;
#pragma unroll
  for (int v = 0; v < 8; ++v) accT.f[v] = 0.f;
  v8f vz = {};
#pragma unroll
  for (int m = 0; m < M_; ++m) {
    const __bf16* rTm = rT + (size_t)m * D_ * R_;
    v8f accM = vz;
#pragma unroll
    for (int kk = 0; kk < 4; ++kk) {
      FragBF b = load_b_fragT(rTm, R_, nt * 16, kk * 32);
      accM = wmma_bf16(afr[kk].v, b.v, accM);
    }
    FragF am; am.v = accM;
#pragma unroll
    for (int v = 0; v < 8; ++v) accT.f[v] += gv[v][m] * am.f[v];
  }
#pragma unroll
  for (int v = 0; v < 8; ++v)
    out[(size_t)(mt * 16 + v + mh) * D_ + nt * 16 + n] = (__bf16)accT.f[v];
}

// ------- causal flash attention: one wave = 16 queries, 32-key blocks -------
__global__ void k_attn(const __bf16* __restrict__ Q, const __bf16* __restrict__ K,
                       const __bf16* __restrict__ Vt, __bf16* __restrict__ O) {
  __shared__ __bf16 plds[4][512];   // per wave: 16 rows x 32 keys
  int wid = threadIdx.x >> 5, lane = threadIdx.x & 31;
  int flat = blockIdx.x * 4 + wid;     // 0..2047
  int qt = flat & 63;
  int bh = flat >> 6;
  int b = bh >> 4, h = bh & 15;
  const __bf16* Qb = Q + (size_t)b * S_ * D_ + h * DH_;
  const __bf16* Kb = K + (size_t)b * S_ * D_ + h * DH_;
  const __bf16* Vtb = Vt + (size_t)bh * DH_ * S_;
  FragBF qa0 = load_a_frag(Qb, D_, qt * 16, 0);
  FragBF qa1 = load_a_frag(Qb, D_, qt * 16, 32);
  int n = lane & 15, mh = (lane >> 4) * 8;
  int kh8 = (lane >> 4) * 8;
  float mrow[8], lrow[8];
  FragF acc[4];
  v8f vz = {};
#pragma unroll
  for (int v = 0; v < 8; ++v) { mrow[v] = -3.0e38f; lrow[v] = 0.f; }
#pragma unroll
  for (int t4 = 0; t4 < 4; ++t4) acc[t4].v = vz;

  int nkb = (qt >> 1) + 1;             // number of 32-key blocks
  for (int kb = 0; kb < nkb; ++kb) {
    int k0 = kb * 32;
    // scores: two 16-key tiles, K(d)=64 via two k-steps each
    FragBF kA0 = load_b_fragT(Kb, D_, k0, 0);
    FragBF kA1 = load_b_fragT(Kb, D_, k0, 32);
    FragBF kB0 = load_b_fragT(Kb, D_, k0 + 16, 0);
    FragBF kB1 = load_b_fragT(Kb, D_, k0 + 16, 32);
    v8f svA = vz, svB = vz;
    svA = wmma_bf16(qa0.v, kA0.v, svA);
    svA = wmma_bf16(qa1.v, kA1.v, svA);
    svB = wmma_bf16(qa0.v, kB0.v, svB);
    svB = wmma_bf16(qa1.v, kB1.v, svB);
    FragF sA; sA.v = svA;
    FragF sB; sB.v = svB;
    bool last = (kb == nkb - 1);
    float alph[8];
#pragma unroll
    for (int v = 0; v < 8; ++v) {
      int q = qt * 16 + v + mh;
      float scA = sA.f[v] * 0.125f;    // 1/sqrt(64)
      float scB = sB.f[v] * 0.125f;
      if (last) {
        if (k0 + n > q)      scA = -3.0e38f;
        if (k0 + 16 + n > q) scB = -3.0e38f;
      }
      float mv = fmaxf(scA, scB);
      mv = fmaxf(mv, __shfl_xor(mv, 1, 16));
      mv = fmaxf(mv, __shfl_xor(mv, 2, 16));
      mv = fmaxf(mv, __shfl_xor(mv, 4, 16));
      mv = fmaxf(mv, __shfl_xor(mv, 8, 16));
      float mn = fmaxf(mrow[v], mv);
      float al = __expf(mrow[v] - mn);
      float pA = __expf(scA - mn);
      float pB = __expf(scB - mn);
      float pr = pA + pB;
      pr += __shfl_xor(pr, 1, 16);
      pr += __shfl_xor(pr, 2, 16);
      pr += __shfl_xor(pr, 4, 16);
      pr += __shfl_xor(pr, 8, 16);
      lrow[v] = lrow[v] * al + pr;
      mrow[v] = mn;
      alph[v] = al;
      plds[wid][(v + mh) * 32 + n]      = (__bf16)pA;   // C-layout -> LDS
      plds[wid][(v + mh) * 32 + 16 + n] = (__bf16)pB;
    }
#pragma unroll
    for (int t4 = 0; t4 < 4; ++t4)
#pragma unroll
      for (int v = 0; v < 8; ++v) acc[t4].f[v] *= alph[v];
    // P back as full K=32 A-fragment
    FragBF pa;
#pragma unroll
    for (int e = 0; e < 16; ++e) {
      int kk = e + ((e >= 8) ? 8 : 0) + kh8;
      pa.h[e] = plds[wid][(lane & 15) * 32 + kk];
    }
#pragma unroll
    for (int t4 = 0; t4 < 4; ++t4) {
      FragBF vb = load_b_fragT(Vtb, S_, t4 * 16, k0);   // BT[N=d][K=key], contiguous
      acc[t4].v = wmma_bf16(pa.v, vb.v, acc[t4].v);
    }
  }
#pragma unroll
  for (int t4 = 0; t4 < 4; ++t4)
#pragma unroll
    for (int v = 0; v < 8; ++v)
      O[(size_t)(b * S_ + qt * 16 + v + mh) * D_ + h * DH_ + t4 * 16 + n] =
          (__bf16)(acc[t4].f[v] / lrow[v]);
}

// ---------------- launcher ----------------
extern "C" void kernel_launch(void* const* d_in, const int* in_sizes, int n_in,
                              void* d_out, int out_size, void* d_ws, size_t ws_size,
                              hipStream_t stream) {
  (void)in_sizes; (void)n_in; (void)out_size; (void)ws_size;
  const float* x    = (const float*)d_in[0];
  const float* qk_f = (const float*)d_in[1];
  const float* qk_r = (const float*)d_in[2];
  const float* v_f  = (const float*)d_in[3];
  const float* v_r  = (const float*)d_in[4];
  const float* Wg_q = (const float*)d_in[5];
  const float* Wg_k = (const float*)d_in[6];
  const float* Wg_v = (const float*)d_in[7];
  const float* W_O  = (const float*)d_in[8];

  char* ws = (char*)d_ws;
  size_t off = 0;
  auto alloc = [&](size_t bytes) -> void* {
    void* p = ws + off;
    off += (bytes + 255) & ~(size_t)255;
    return p;
  };
  __bf16* xb     = (__bf16*)alloc((size_t)BS_ * D_ * 2);
  __bf16* WgT    = (__bf16*)alloc((size_t)3 * NN_ * D_ * 2);
  __bf16* qk_fT  = (__bf16*)alloc((size_t)M_ * R_ * D_ * 2);
  __bf16* v_fT   = (__bf16*)alloc((size_t)M_ * R_ * D_ * 2);
  __bf16* qk_rT  = (__bf16*)alloc((size_t)M_ * D_ * R_ * 2);
  __bf16* v_rT   = (__bf16*)alloc((size_t)M_ * D_ * R_ * 2);
  __bf16* W_OT   = (__bf16*)alloc((size_t)D_ * D_ * 2);
  float*  logits = (float*)alloc((size_t)3 * BS_ * NN_ * 4);
  float*  gates  = (float*)alloc((size_t)3 * BS_ * M_ * 4);
  float*  allhqk = (float*)alloc((size_t)M_ * BS_ * R_ * 4);
  float*  allhv  = (float*)alloc((size_t)M_ * BS_ * R_ * 4);
  __bf16* hQ     = (__bf16*)alloc((size_t)BS_ * R_ * 2);
  __bf16* hK     = (__bf16*)alloc((size_t)BS_ * R_ * 2);
  __bf16* hV     = (__bf16*)alloc((size_t)BS_ * R_ * 2);
  __bf16* Qb     = (__bf16*)alloc((size_t)BS_ * D_ * 2);
  __bf16* Kb     = (__bf16*)alloc((size_t)BS_ * D_ * 2);
  __bf16* Vb     = (__bf16*)alloc((size_t)BS_ * D_ * 2);
  __bf16* Vt     = (__bf16*)alloc((size_t)BS_ * D_ * 2);
  __bf16* attno  = (__bf16*)alloc((size_t)BS_ * D_ * 2);

  auto cdiv = [](int a, int b) { return (a + b - 1) / b; };

  // ---- conversions ----
  k_f2bf<<<cdiv(BS_ * D_, 256), 256, 0, stream>>>(x, xb, BS_ * D_);
  k_f2bf_t<<<cdiv(D_ * NN_, 256), 256, 0, stream>>>(Wg_q, WgT + 0 * (size_t)NN_ * D_, 1, D_, NN_);
  k_f2bf_t<<<cdiv(D_ * NN_, 256), 256, 0, stream>>>(Wg_k, WgT + 1 * (size_t)NN_ * D_, 1, D_, NN_);
  k_f2bf_t<<<cdiv(D_ * NN_, 256), 256, 0, stream>>>(Wg_v, WgT + 2 * (size_t)NN_ * D_, 1, D_, NN_);
  k_f2bf_t<<<cdiv(M_ * D_ * R_, 256), 256, 0, stream>>>(qk_f, qk_fT, M_, D_, R_);
  k_f2bf_t<<<cdiv(M_ * D_ * R_, 256), 256, 0, stream>>>(v_f,  v_fT,  M_, D_, R_);
  k_f2bf_t<<<cdiv(M_ * R_ * D_, 256), 256, 0, stream>>>(qk_r, qk_rT, M_, R_, D_);
  k_f2bf_t<<<cdiv(M_ * R_ * D_, 256), 256, 0, stream>>>(v_r,  v_rT,  M_, R_, D_);
  k_f2bf_t<<<cdiv(D_ * D_, 256), 256, 0, stream>>>(W_O, W_OT, 1, D_, D_);

  // ---- router logits (z=3) + gates ----
  k_gemm<<<dim3(NN_ / 64, BS_ / 64, 3), 128, 0, stream>>>(
      xb, WgT, logits, NN_, D_, (long long)NN_ * D_, (long long)BS_ * NN_);
  k_gates<<<(3 * BS_) / 4, 128, 0, stream>>>(logits, gates);
  float* gQ = gates;
  float* gK = gates + (size_t)BS_ * M_;
  float* gV = gates + 2 * (size_t)BS_ * M_;

  // ---- all_h = x @ f[m] (z=8), shared between Q and K ----
  k_gemm<<<dim3(R_ / 64, BS_ / 64, M_), 128, 0, stream>>>(
      xb, qk_fT, allhqk, R_, D_, (long long)R_ * D_, (long long)BS_ * R_);
  k_gemm<<<dim3(R_ / 64, BS_ / 64, M_), 128, 0, stream>>>(
      xb, v_fT, allhv, R_, D_, (long long)R_ * D_, (long long)BS_ * R_);

  // ---- gate-weighted combine -> h (bf16) ----
  k_h<<<cdiv(BS_ * R_, 256), 256, 0, stream>>>(allhqk, gQ, hQ);
  k_h<<<cdiv(BS_ * R_, 256), 256, 0, stream>>>(allhqk, gK, hK);
  k_h<<<cdiv(BS_ * R_, 256), 256, 0, stream>>>(allhv,  gV, hV);

  // ---- restore -> Q, K, V (bf16 [BS_, D_]) ----
  k_restore<<<dim3(D_ / 16, BS_ / 64), 128, 0, stream>>>(hQ, gQ, qk_rT, Qb);
  k_restore<<<dim3(D_ / 16, BS_ / 64), 128, 0, stream>>>(hK, gK, qk_rT, Kb);
  k_restore<<<dim3(D_ / 16, BS_ / 64), 128, 0, stream>>>(hV, gV, v_rT,  Vb);

  // ---- V transpose for contiguous P@V B-fragments ----
  k_vtrans<<<cdiv(BS_ * D_, 256), 256, 0, stream>>>(Vb, Vt);

  // ---- causal flash attention ----
  k_attn<<<(B_ * H_ * (S_ / 16)) / 4, 128, 0, stream>>>(Qb, Kb, Vt, attno);

  // ---- output projection -> d_out (fp32) ----
  k_gemm<<<dim3(D_ / 64, BS_ / 64, 1), 128, 0, stream>>>(
      attno, W_OT, (float*)d_out, D_, D_, 0, 0);
}
